// GCN_128849018930
// MI455X (gfx1250) — matrix operand order repaired and checked by reading, not compiled, not run on hardware
//
#include <hip/hip_runtime.h>
#include <hip/hip_bf16.h>

#define IN_DIM 512
#define HID    256
#define OUTD   64

typedef __bf16 bf16_t;
typedef __attribute__((ext_vector_type(16))) __bf16 v16bf;
typedef __attribute__((ext_vector_type(8)))  __bf16 v8bf;
typedef __attribute__((ext_vector_type(8)))  float  v8f;

// ---------------------------------------------------------------------------
// Pack a row-major fp32 weight [K][N] into per-lane WMMA B fragments (bf16).
// Fragment layout (v_wmma_f32_16x16x32_bf16 B-matrix, wave32):
//   lane < 16 : N = ntile*16 + lane,     element j -> K = ktile*32 + (j&7) + (j>>3)*16
//   lane >= 16: N = ntile*16 + lane-16,  element j -> K = ktile*32 + 8 + (j&7) + (j>>3)*16
// Packed index = ((ktile*(N/16) + ntile)*32 + lane)*16 + j
// ---------------------------------------------------------------------------
__global__ void pack_w_kernel(const float* __restrict__ w, bf16_t* __restrict__ out,
                              int K, int N) {
    int idx = blockIdx.x * blockDim.x + threadIdx.x;
    int total = K * N;
    if (idx >= total) return;
    int j    = idx & 15;
    int lane = (idx >> 4) & 31;
    int tile = idx >> 9;
    int ntiles = N >> 4;
    int ktile = tile / ntiles;
    int ntile = tile - ktile * ntiles;
    int n = ntile * 16 + (lane & 15);
    int k = ktile * 32 + ((lane >> 4) ? 8 : 0) + (j & 7) + ((j >> 3) * 16);
    out[idx] = (bf16_t)w[k * N + n];
}

// Build a v16bf A fragment for one 16x32 bf16 tile (ISA 16-bit A layout):
// lane<16: K chunks [kk+0..7],[kk+16..23]; lane>=16: [kk+8..15],[kk+24..31].
__device__ __forceinline__ v16bf load_a_frag(const bf16_t* row, int kk, int khalf) {
    v8bf a0 = *(const v8bf*)(row + kk + khalf * 8);
    v8bf a1 = *(const v8bf*)(row + kk + khalf * 8 + 16);
    return __builtin_shufflevector(a0, a1, 0,1,2,3,4,5,6,7,8,9,10,11,12,13,14,15);
}

// ---------------------------------------------------------------------------
// GEMM1: pre1[N,256] = X[N,512] @ W1   (W1 pre-packed bf16 fragments)
// 512 threads = 16 waves; block computes 32 rows x 256 cols. Wave w covers
// cols [16w,16w+16) and TWO 16-row tiles -> one B fragment feeds 2 WMMAs.
// ---------------------------------------------------------------------------
__global__ __launch_bounds__(512)
void gemm_xw1_kernel(const float* __restrict__ X, const bf16_t* __restrict__ Wp,
                     float* __restrict__ out, int nNodes) {
    __shared__ bf16_t As[32][IN_DIM + 8];   // 33.3 KB, padded rows (bank spread)

    const int tid     = threadIdx.x;
    const int lane    = tid & 31;
    const int wave    = tid >> 5;           // 0..15
    const int rowBase = blockIdx.x * 32;
    const bool full   = (rowBase + 32 <= nNodes);

    // Stage + convert: thread t -> row t/16, cols [(t%16)*32, +32)
    {
        int r  = tid >> 4;
        int c  = (tid & 15) * 32;
        int rr = rowBase + r; if (rr >= nNodes) rr = nNodes - 1;
        const float4* src = (const float4*)(X + (size_t)rr * IN_DIM + c);
#pragma unroll
        for (int q = 0; q < 8; ++q) {
            float4 f = src[q];
            As[r][c + q * 4 + 0] = (bf16_t)f.x;
            As[r][c + q * 4 + 1] = (bf16_t)f.y;
            As[r][c + q * 4 + 2] = (bf16_t)f.z;
            As[r][c + q * 4 + 3] = (bf16_t)f.w;
        }
    }
    __syncthreads();

    const int m     = lane & 15;
    const int khalf = lane >> 4;            // 0 or 1
    const v16bf* Bp = (const v16bf*)Wp;

    v8f acc0 = {}, acc1 = {};
#pragma unroll 4
    for (int kt = 0; kt < IN_DIM / 32; ++kt) {
        const int kk = kt * 32;
        v16bf a0 = load_a_frag(&As[m][0],      kk, khalf);
        v16bf a1 = load_a_frag(&As[m + 16][0], kk, khalf);
        v16bf b  = Bp[(size_t)(kt * (HID / 16) + wave) * 32 + lane];
        acc0 = __builtin_amdgcn_wmma_f32_16x16x32_bf16(false, a0, false, b, (short)0, acc0, false, false);
        acc1 = __builtin_amdgcn_wmma_f32_16x16x32_bf16(false, a1, false, b, (short)0, acc1, false, false);
    }

    // C/D layout: VGPR r -> (M = khalf*8 + r, N = lane%16)
    const int col = wave * 16 + (lane & 15);
    float* p0 = out + (size_t)(rowBase + khalf * 8) * HID + col;
    if (full) {
#pragma unroll
        for (int r = 0; r < 8; ++r) p0[(size_t)r * HID] = acc0[r];
#pragma unroll
        for (int r = 0; r < 8; ++r) p0[(size_t)(r + 16) * HID] = acc1[r];
    } else {
#pragma unroll
        for (int r = 0; r < 8; ++r) {
            int row = rowBase + khalf * 8 + r;
            if (row < nNodes)      out[(size_t)row * HID + col] = acc0[r];
            if (row + 16 < nNodes) out[(size_t)(row + 16) * HID + col] = acc1[r];
        }
    }
}

// ---------------------------------------------------------------------------
// GEMM2: pre2[N,64] = relu(h)[N,256] @ W2   (relu fused into LDS staging)
// 128 threads = 4 waves; block = 32 rows x 64 cols, two row tiles per wave.
// ---------------------------------------------------------------------------
__global__ __launch_bounds__(128)
void gemm_hw2_kernel(const float* __restrict__ H, const bf16_t* __restrict__ Wp,
                     float* __restrict__ out, int nNodes) {
    __shared__ bf16_t As[32][HID + 8];      // 16.9 KB

    const int tid     = threadIdx.x;
    const int lane    = tid & 31;
    const int wave    = tid >> 5;           // 0..3
    const int rowBase = blockIdx.x * 32;
    const bool full   = (rowBase + 32 <= nNodes);

    // Stage with ReLU: thread t -> row t/4, cols [(t%4)*64, +64)
    {
        int r  = tid >> 2;
        int c  = (tid & 3) * 64;
        int rr = rowBase + r; if (rr >= nNodes) rr = nNodes - 1;
        const float4* src = (const float4*)(H + (size_t)rr * HID + c);
#pragma unroll
        for (int q = 0; q < 16; ++q) {
            float4 f = src[q];
            As[r][c + q * 4 + 0] = (bf16_t)fmaxf(f.x, 0.f);
            As[r][c + q * 4 + 1] = (bf16_t)fmaxf(f.y, 0.f);
            As[r][c + q * 4 + 2] = (bf16_t)fmaxf(f.z, 0.f);
            As[r][c + q * 4 + 3] = (bf16_t)fmaxf(f.w, 0.f);
        }
    }
    __syncthreads();

    const int m     = lane & 15;
    const int khalf = lane >> 4;
    const v16bf* Bp = (const v16bf*)Wp;

    v8f acc0 = {}, acc1 = {};
#pragma unroll
    for (int kt = 0; kt < HID / 32; ++kt) {
        const int kk = kt * 32;
        v16bf a0 = load_a_frag(&As[m][0],      kk, khalf);
        v16bf a1 = load_a_frag(&As[m + 16][0], kk, khalf);
        v16bf b  = Bp[(size_t)(kt * (OUTD / 16) + wave) * 32 + lane];
        acc0 = __builtin_amdgcn_wmma_f32_16x16x32_bf16(false, a0, false, b, (short)0, acc0, false, false);
        acc1 = __builtin_amdgcn_wmma_f32_16x16x32_bf16(false, a1, false, b, (short)0, acc1, false, false);
    }

    const int col = wave * 16 + (lane & 15);
    float* p0 = out + (size_t)(rowBase + khalf * 8) * OUTD + col;
    if (full) {
#pragma unroll
        for (int r = 0; r < 8; ++r) p0[(size_t)r * OUTD] = acc0[r];
#pragma unroll
        for (int r = 0; r < 8; ++r) p0[(size_t)(r + 16) * OUTD] = acc1[r];
    } else {
#pragma unroll
        for (int r = 0; r < 8; ++r) {
            int row = rowBase + khalf * 8 + r;
            if (row < nNodes)      out[(size_t)row * OUTD + col] = acc0[r];
            if (row + 16 < nNodes) out[(size_t)(row + 16) * OUTD + col] = acc1[r];
        }
    }
}

// ---------------------------------------------------------------------------
// SpMM, D=256: one wave per edge. out[src] += val * dense[dst].
// Coalesced float4 row reads (L2-resident), hardware f32 atomic scatter-add.
// ---------------------------------------------------------------------------
__global__ __launch_bounds__(256)
void spmm256_kernel(const int* __restrict__ src, const int* __restrict__ dst,
                    const float* __restrict__ val, const float* __restrict__ dense,
                    float* __restrict__ out, int nE) {
    int gw   = (blockIdx.x * blockDim.x + threadIdx.x) >> 5;
    int lane = threadIdx.x & 31;
    if (gw >= nE) return;
    int   s = src[gw];
    int   d = dst[gw];
    float v = val[gw];
    const float4* drow = (const float4*)(dense + (size_t)d * HID);
    float*        orow = out + (size_t)s * HID;
#pragma unroll
    for (int i = 0; i < 2; ++i) {
        float4 x = drow[lane + 32 * i];
        int b = (lane + 32 * i) * 4;
        unsafeAtomicAdd(orow + b + 0, v * x.x);
        unsafeAtomicAdd(orow + b + 1, v * x.y);
        unsafeAtomicAdd(orow + b + 2, v * x.z);
        unsafeAtomicAdd(orow + b + 3, v * x.w);
    }
}

// SpMM, D=64: one wave per edge, float2 per lane.
__global__ __launch_bounds__(256)
void spmm64_kernel(const int* __restrict__ src, const int* __restrict__ dst,
                   const float* __restrict__ val, const float* __restrict__ dense,
                   float* __restrict__ out, int nE) {
    int gw   = (blockIdx.x * blockDim.x + threadIdx.x) >> 5;
    int lane = threadIdx.x & 31;
    if (gw >= nE) return;
    int   s = src[gw];
    int   d = dst[gw];
    float v = val[gw];
    const float2* drow = (const float2*)(dense + (size_t)d * OUTD);
    float*        orow = out + (size_t)s * OUTD;
    float2 x = drow[lane];
    unsafeAtomicAdd(orow + lane * 2 + 0, v * x.x);
    unsafeAtomicAdd(orow + lane * 2 + 1, v * x.y);
}

__global__ void zero_kernel(float4* __restrict__ p, size_t n4) {
    size_t i      = blockIdx.x * (size_t)blockDim.x + threadIdx.x;
    size_t stride = (size_t)gridDim.x * blockDim.x;
    float4 z = make_float4(0.f, 0.f, 0.f, 0.f);
    for (; i < n4; i += stride) p[i] = z;
}

// ---------------------------------------------------------------------------
extern "C" void kernel_launch(void* const* d_in, const int* in_sizes, int n_in,
                              void* d_out, int out_size, void* d_ws, size_t ws_size,
                              hipStream_t stream) {
    const float* x     = (const float*)d_in[0];
    const int*   esrc  = (const int*)  d_in[1];
    const int*   edst  = (const int*)  d_in[2];
    const float* evalv = (const float*)d_in[3];
    const float* w1    = (const float*)d_in[4];
    const float* w2    = (const float*)d_in[5];
    float*       out   = (float*)d_out;

    const int nNodes = in_sizes[0] / IN_DIM;   // 100000
    const int nE     = in_sizes[1];            // 3200000

    // Workspace layout:
    //   [0, 256K)        packed W1 (bf16)
    //   [512K, 544K)     packed W2 (bf16)
    //   [1M, 1M+N*256*4) bufA : pre1, later reused for pre2
    //   [.., +N*256*4)   bufB : h accumulator
    char*   ws   = (char*)d_ws;
    bf16_t* w1p  = (bf16_t*)ws;
    bf16_t* w2p  = (bf16_t*)(ws + (512 << 10));
    float*  bufA = (float*)(ws + (1 << 20));
    float*  bufB = bufA + (size_t)nNodes * HID;

    // 1) Pack weights into WMMA fragment layout (bf16)
    pack_w_kernel<<<(IN_DIM * HID + 255) / 256, 256, 0, stream>>>(w1, w1p, IN_DIM, HID);
    pack_w_kernel<<<(HID * OUTD + 255) / 256, 256, 0, stream>>>(w2, w2p, HID, OUTD);

    // 2) pre1 = X @ W1
    gemm_xw1_kernel<<<(nNodes + 31) / 32, 512, 0, stream>>>(x, w1p, bufA, nNodes);

    // 3) h = A @ pre1 (scatter-add; zero the accumulator first)
    size_t hElems = (size_t)nNodes * HID;
    zero_kernel<<<2048, 256, 0, stream>>>((float4*)bufB, hElems / 4);
    spmm256_kernel<<<(nE + 7) / 8, 256, 0, stream>>>(esrc, edst, evalv, bufA, bufB, nE);

    // 4) pre2 = relu(h) @ W2   (pre1 dead -> reuse bufA)
    gemm_hw2_kernel<<<(nNodes + 31) / 32, 128, 0, stream>>>(bufB, w2p, bufA, nNodes);

    // 5) logits = A @ pre2 into d_out (zero first; atomics accumulate)
    size_t oElems = (size_t)nNodes * OUTD;
    zero_kernel<<<2048, 256, 0, stream>>>((float4*)out, oElems / 4);
    spmm64_kernel<<<(nE + 7) / 8, 256, 0, stream>>>(esrc, edst, evalv, bufA, out, nE);
}
